// EncoderLayer_8985071583693
// MI455X (gfx1250) — matrix operand order repaired and checked by reading, not compiled, not run on hardware
//
#include <hip/hip_runtime.h>

typedef __attribute__((ext_vector_type(16))) __bf16 v16bf;
typedef __attribute__((ext_vector_type(8)))  __bf16 v8bf;
typedef __attribute__((ext_vector_type(4)))  __bf16 v4bf;
typedef __attribute__((ext_vector_type(8)))  float  v8f;
typedef __attribute__((ext_vector_type(4)))  float  v4f;

static constexpr int B_ = 4, L_ = 2048, D_ = 1024, F_ = 4096, H_ = 16, E_ = 64;

// ---------- WMMA helpers (CDNA5 16x16x32 bf16, fp32 accumulate) ----------

__device__ __forceinline__ v8f wmma_bf16(v16bf a, v16bf b, v8f c) {
  // 8 args: (neg_a, A, neg_b, B, c_mod, C, reuse_a, reuse_b)
  return __builtin_amdgcn_wmma_f32_16x16x32_bf16(false, a, false, b, (short)0, c,
                                                 false, false);
}

__device__ __forceinline__ v8f zero8f() { return v8f{0, 0, 0, 0, 0, 0, 0, 0}; }

// A-matrix fragment (16x32 bf16, M x K), source row-major with leading dim ld.
// Lane l (0..15): row M=l, K = 0..7 (f[0..7]) and 16..23 (f[8..15])
// Lane l (16..31): row M=l-16, K = 8..15 and 24..31
__device__ __forceinline__ v16bf load_fragA(const __bf16* p, int ld) {
  const int lane = threadIdx.x & 31;
  const int row  = lane & 15;
  const int kb   = (lane >> 4) << 3; // 0 or 8
  const __bf16* r0 = p + (size_t)row * ld + kb;
  v8bf lo = *(const v8bf*)(r0);
  v8bf hi = *(const v8bf*)(r0 + 16);
  v16bf f;
#pragma unroll
  for (int i = 0; i < 8; ++i) { f[i] = lo[i]; f[i + 8] = hi[i]; }
  return f;
}

// B-matrix fragment (32x16 bf16, K x N), sourced from the TRANSPOSE of B
// stored row-major as N x K with leading dim ld (i.e. a weight matrix [N,K]).
// Lane l (0..15): col N=l, K = 0..7 and 16..23; lanes 16..31: K = 8..15 and 24..31.
__device__ __forceinline__ v16bf load_fragB(const __bf16* p, int ld) {
  const int lane = threadIdx.x & 31;
  const int n  = lane & 15;
  const int kb = (lane >> 4) << 3; // 0 or 8
  const __bf16* r0 = p + (size_t)n * ld + kb;
  v8bf lo = *(const v8bf*)(r0);
  v8bf hi = *(const v8bf*)(r0 + 16);
  v16bf f;
#pragma unroll
  for (int i = 0; i < 8; ++i) { f[i] = lo[i]; f[i + 8] = hi[i]; }
  return f;
}

// ---------- CDNA5 async global->LDS copy (ASYNCcnt path, inline asm) ----------
// VDST = LDS byte offset (flat LDS pointer's low 32 bits ARE the LDS offset,
// ISA 10.2 aperture table), VADDR = 64-bit global address, saddr = off.

__device__ __forceinline__ void async_copy16(const __bf16* g, __bf16* l) {
  unsigned lds_off = (unsigned)(unsigned long long)l;
  asm volatile("global_load_async_to_lds_b128 %0, %1, off"
               :
               : "v"(lds_off), "v"(g)
               : "memory");
}

__device__ __forceinline__ void wait_async0() {
#if __has_builtin(__builtin_amdgcn_s_wait_asynccnt)
  __builtin_amdgcn_s_wait_asynccnt(0);
#else
  asm volatile("s_wait_asynccnt 0" ::: "memory");
#endif
}

// ---------- fp32 -> bf16 cast (vectorized x4) ----------

__global__ void cast_f32_bf16(const float* __restrict__ in, __bf16* __restrict__ out,
                              size_t n4) {
  size_t i  = (size_t)blockIdx.x * blockDim.x + threadIdx.x;
  size_t st = (size_t)gridDim.x * blockDim.x;
  for (; i < n4; i += st) {
    v4f v = ((const v4f*)in)[i];
    v4bf o;
#pragma unroll
    for (int k = 0; k < 4; ++k) o[k] = (__bf16)v[k];
    ((v4bf*)out)[i] = o;
  }
}

// ---------- Flash attention: per (b,h), Q=K=V = heads of x ----------
// grid.x = B*H*16; 8 waves/block, each wave owns one 16-row query tile.
__global__ __launch_bounds__(256) void attn_kernel(const __bf16* __restrict__ xb,
                                                   float* __restrict__ attn) {
  __shared__ __align__(16) __bf16 Ks[32 * 64];     // K chunk: 32 keys x 64 dims
  __shared__ __align__(16) __bf16 Vt[64 * 32];     // V chunk transposed: dim x key
  __shared__ __align__(16) __bf16 Pb[8][16 * 32];  // per-wave P relayout scratch

  const int t = threadIdx.x, w = t >> 5, lane = t & 31;
  const int bh = blockIdx.x >> 4;   // (b,h)
  const int qg = blockIdx.x & 15;
  const int b  = bh >> 4;           // H_ == 16
  const int h  = bh & 15;
  const int qt = qg * 8 + w;        // query tile 0..127

  const __bf16* xh = xb + (size_t)b * L_ * D_ + h * E_;
  const __bf16* qp = xh + (size_t)(qt * 16) * D_;
  const v16bf qa0 = load_fragA(qp, D_);       // Q dims 0..31
  const v16bf qa1 = load_fragA(qp + 32, D_);  // Q dims 32..63

  v16bf onesb;                                // 32x16 all-ones B operand
#pragma unroll
  for (int i = 0; i < 16; ++i) onesb[i] = (__bf16)1.0f;

  v8f o[4];
#pragma unroll
  for (int g = 0; g < 4; ++g) o[g] = zero8f();
  float rm[8], rs[8];
#pragma unroll
  for (int r = 0; r < 8; ++r) { rm[r] = -3.0e38f; rs[r] = 0.f; }

  const int   half  = lane >> 4;
  const int   n0    = lane & 15;
  const float scale = 0.125f; // 1/sqrt(64)

  for (int sc = 0; sc < L_ / 32; ++sc) {
    __syncthreads();
    { // stage 32 key/value rows; K direct, V transposed (K == V == x heads)
      const int s  = t >> 3;
      const int e8 = (t & 7) << 3;
      v8bf d = *(const v8bf*)(xh + (size_t)(sc * 32 + s) * D_ + e8);
      *(v8bf*)&Ks[s * 64 + e8] = d;
#pragma unroll
      for (int i = 0; i < 8; ++i) Vt[(e8 + i) * 32 + s] = d[i];
    }
    __syncthreads();

    // scores: 16 queries x 32 keys, E=64 split in two K=32 WMMAs per key group
    v8f sf[2];
#pragma unroll
    for (int g = 0; g < 2; ++g) {
      v16bf kb0 = load_fragB(&Ks[(g * 16) * 64 + 0], 64);
      v16bf kb1 = load_fragB(&Ks[(g * 16) * 64 + 32], 64);
      v8f acc = zero8f();
      acc = wmma_bf16(qa0, kb0, acc);
      acc = wmma_bf16(qa1, kb1, acc);
      sf[g] = acc;
    }

    // online softmax: cross-lane max (ds shuffles), sum via ones-WMMA below
    float p0[8], p1[8], corr[8];
#pragma unroll
    for (int r = 0; r < 8; ++r) {
      float s0 = sf[0][r] * scale;
      float s1 = sf[1][r] * scale;
      float mx = fmaxf(s0, s1);
#pragma unroll
      for (int off = 1; off < 16; off <<= 1) mx = fmaxf(mx, __shfl_xor(mx, off, 16));
      float mn = fmaxf(rm[r], mx);
      corr[r]  = __expf(rm[r] - mn);
      rm[r]    = mn;
      p0[r]    = __expf(s0 - mn);
      p1[r]    = __expf(s1 - mn);
#pragma unroll
      for (int g = 0; g < 4; ++g) o[g][r] *= corr[r];
    }

    // relayout P (C-frag) -> A-frag through wave-private LDS
#pragma unroll
    for (int r = 0; r < 8; ++r) {
      int m = r + 8 * half;
      Pb[w][m * 32 + n0]      = (__bf16)p0[r];
      Pb[w][m * 32 + n0 + 16] = (__bf16)p1[r];
    }
    v16bf pa = load_fragA(&Pb[w][0], 32);

    // row sums of P via WMMA against ones (replicated into every lane)
    v8f rsum = wmma_bf16(pa, onesb, zero8f());
#pragma unroll
    for (int r = 0; r < 8; ++r) rs[r] = rs[r] * corr[r] + rsum[r];

#pragma unroll
    for (int g = 0; g < 4; ++g) {
      v16bf vb = load_fragB(&Vt[(g * 16) * 32], 32);
      o[g] = wmma_bf16(pa, vb, o[g]);
    }
  }

  float* op = attn + ((size_t)b * L_ + qt * 16) * D_ + h * E_;
#pragma unroll
  for (int g = 0; g < 4; ++g)
#pragma unroll
    for (int r = 0; r < 8; ++r) {
      int m = r + 8 * half;
      op[(size_t)m * D_ + g * 16 + n0] = o[g][r] / rs[r];
    }
}

// ---------- residual + LayerNorm (one row of D=1024 per block) ----------

__global__ __launch_bounds__(256) void resid_ln(const float* __restrict__ xa,
                                                const float* __restrict__ xr,
                                                const float* __restrict__ gamma,
                                                const float* __restrict__ beta,
                                                float* __restrict__ outf,
                                                __bf16* __restrict__ outb) {
  __shared__ float red[256];
  const int row = blockIdx.x, t = threadIdx.x;
  const float* pa = xa + (size_t)row * D_;
  const float* pr = xr + (size_t)row * D_;
  float v[4];
  float s = 0.f;
#pragma unroll
  for (int i = 0; i < 4; ++i) { v[i] = pa[t + i * 256] + pr[t + i * 256]; s += v[i]; }
  red[t] = s;
  __syncthreads();
  for (int o = 128; o > 0; o >>= 1) { if (t < o) red[t] += red[t + o]; __syncthreads(); }
  float mu = red[0] * (1.f / D_);
  __syncthreads();
  float q = 0.f;
#pragma unroll
  for (int i = 0; i < 4; ++i) { float d = v[i] - mu; q += d * d; }
  red[t] = q;
  __syncthreads();
  for (int o = 128; o > 0; o >>= 1) { if (t < o) red[t] += red[t + o]; __syncthreads(); }
  float inv = rsqrtf(red[0] * (1.f / D_) + 1e-5f);
  float* po = outf + (size_t)row * D_;
#pragma unroll
  for (int i = 0; i < 4; ++i) {
    int c = t + i * 256;
    float y = (v[i] - mu) * inv * gamma[c] + beta[c];
    po[c] = y;
    if (outb) outb[(size_t)row * D_ + c] = (__bf16)y;
  }
}

// ---------- bf16 GEMM: C[M,N] = A[M,K] x W[N,K]^T + bias, optional relu ----------
// 256x128 block tile, 8 waves of 64x64, K staged in 32-wide LDS chunks with
// async-to-LDS double buffering (ASYNCcnt overlaps DMA with WMMA compute).
template <bool RELU, bool OUTBF>
__global__ __launch_bounds__(256) void gemm_bf16_k(const __bf16* __restrict__ A,
                                                   const __bf16* __restrict__ W,
                                                   const float* __restrict__ bias,
                                                   float* __restrict__ Cf,
                                                   __bf16* __restrict__ Cb,
                                                   int M, int N, int K) {
  __shared__ __align__(16) __bf16 As[2][256 * 32];
  __shared__ __align__(16) __bf16 Ws[2][128 * 32];
  const int t = threadIdx.x, lane = t & 31;
  const int w  = t >> 5;
  const int wr = w >> 1, wc = w & 1; // 4x2 wave grid, 64x64 per wave

  const int bm = blockIdx.x, bn = blockIdx.y;
  const __bf16* Ab = A + (size_t)(bm * 256) * K;
  const __bf16* Wb = W + (size_t)(bn * 128) * K;

  v8f c[4][4];
#pragma unroll
  for (int i = 0; i < 4; ++i)
#pragma unroll
    for (int j = 0; j < 4; ++j) c[i][j] = zero8f();

  auto stage = [&](int buf, int kc) {
#pragma unroll
    for (int i = 0; i < 4; ++i) { // A: 256x32 = 1024 16B vectors, 4/thread
      int j = t + i * 256;
      int row = j >> 2, c8 = (j & 3) << 3;
      async_copy16(Ab + (size_t)row * K + kc + c8, &As[buf][row * 32 + c8]);
    }
#pragma unroll
    for (int i = 0; i < 2; ++i) { // W: 128x32 = 512 16B vectors, 2/thread
      int j = t + i * 256;
      int row = j >> 2, c8 = (j & 3) << 3;
      async_copy16(Wb + (size_t)row * K + kc + c8, &Ws[buf][row * 32 + c8]);
    }
  };

  stage(0, 0);
  int buf = 0;
  for (int kc = 0; kc < K; kc += 32, buf ^= 1) {
    wait_async0();      // own async copies into `buf` complete
    __syncthreads();    // everyone's copies visible; prev compute done
    if (kc + 32 < K) stage(buf ^ 1, kc + 32); // prefetch next chunk (DMA)

    v16bf bf[4];
#pragma unroll
    for (int j = 0; j < 4; ++j)
      bf[j] = load_fragB(&Ws[buf][(wc * 64 + j * 16) * 32], 32);
#pragma unroll
    for (int i = 0; i < 4; ++i) {
      v16bf af = load_fragA(&As[buf][(wr * 64 + i * 16) * 32], 32);
#pragma unroll
      for (int j = 0; j < 4; ++j) c[i][j] = wmma_bf16(af, bf[j], c[i][j]);
    }
  }

  const int half = lane >> 4, n0 = lane & 15;
#pragma unroll
  for (int i = 0; i < 4; ++i)
#pragma unroll
    for (int j = 0; j < 4; ++j)
#pragma unroll
      for (int r = 0; r < 8; ++r) {
        int row = bm * 256 + wr * 64 + i * 16 + r + 8 * half;
        int col = bn * 128 + wc * 64 + j * 16 + n0;
        float v = c[i][j][r] + bias[col];
        if (RELU) v = fmaxf(v, 0.f);
        if (OUTBF) Cb[(size_t)row * N + col] = (__bf16)v;
        else       Cf[(size_t)row * N + col] = v;
      }
}

// ---------- launch ----------

extern "C" void kernel_launch(void* const* d_in, const int* in_sizes, int n_in,
                              void* d_out, int out_size, void* d_ws, size_t ws_size,
                              hipStream_t stream) {
  const float* x   = (const float*)d_in[0];
  const float* w1  = (const float*)d_in[1];
  const float* b1  = (const float*)d_in[2];
  const float* w2  = (const float*)d_in[3];
  const float* b2  = (const float*)d_in[4];
  const float* g1  = (const float*)d_in[5];
  const float* be1 = (const float*)d_in[6];
  const float* g2  = (const float*)d_in[7];
  const float* be2 = (const float*)d_in[8];

  const size_t BLD = (size_t)B_ * L_ * D_; // 8388608
  const size_t FD  = (size_t)F_ * D_;      // 4194304
  const size_t BLF = (size_t)B_ * L_ * F_; // 33554432

  char* p = (char*)d_ws;
  __bf16* xb   = (__bf16*)p; p += BLD * 2;
  float*  attn = (float*)p;  p += BLD * 4;
  float*  x1f  = (float*)p;  p += BLD * 4;
  __bf16* x1b  = (__bf16*)p; p += BLD * 2;
  __bf16* w1b  = (__bf16*)p; p += FD * 2;
  __bf16* w2b  = (__bf16*)p; p += FD * 2;
  __bf16* yb   = (__bf16*)p; p += BLF * 2;
  float*  z    = (float*)p;  p += BLD * 4;

  cast_f32_bf16<<<1024, 256, 0, stream>>>(x, xb, BLD / 4);
  cast_f32_bf16<<<512, 256, 0, stream>>>(w1, w1b, FD / 4);
  cast_f32_bf16<<<512, 256, 0, stream>>>(w2, w2b, FD / 4);

  attn_kernel<<<B_ * H_ * 16, 256, 0, stream>>>(xb, attn);

  resid_ln<<<B_ * L_, 256, 0, stream>>>(x, attn, g1, be1, x1f, x1b);

  gemm_bf16_k<true, true><<<dim3(8192 / 256, 4096 / 128), 256, 0, stream>>>(
      x1b, w1b, b1, nullptr, yb, 8192, 4096, 1024);
  gemm_bf16_k<false, false><<<dim3(8192 / 256, 1024 / 128), 256, 0, stream>>>(
      yb, w2b, b2, z, nullptr, 8192, 1024, 4096);

  resid_ln<<<B_ * L_, 256, 0, stream>>>(x1f, z, g2, be2, (float*)d_out, nullptr);
}
